// PointerSelectorV1_39737037423364
// MI455X (gfx1250) — compile-verified, matrix operand order checked by reading, not run on hardware
//
#include <hip/hip_runtime.h>
#include <stdint.h>

#define D      768
#define KC     32
#define BATCH  4096
#define SHOT   6
#define TEMP   0.1f
#define AST    776   // padded LDS row stride (ushorts): 1552 B rows, 16B-aligned, bank-spread
#define ROWS   64    // rows per workgroup in the MLP kernel (4 M-tiles)
#define MT     4     // M-tiles per workgroup
#define NPW    3     // N-tiles per wave (48 tiles / 16 waves)
#define MLP_THREADS 512

typedef __attribute__((ext_vector_type(16))) __bf16 v16bf;
typedef __attribute__((ext_vector_type(8)))  float  v8f;
typedef unsigned short u16;
typedef unsigned int   u32;
typedef unsigned long long u64;

__device__ __forceinline__ float bf2f(u16 h) {
    union { u32 u; float f; } x; x.u = ((u32)h) << 16; return x.f;
}
__device__ __forceinline__ u16 f2bf(float f) {
    union { float f; u32 u; } x; x.f = f;
    u32 r = x.u + 0x7FFFu + ((x.u >> 16) & 1u);
    return (u16)(r >> 16);
}

__device__ __forceinline__ void wait_asynccnt0() {
#if __has_builtin(__builtin_amdgcn_s_wait_asynccnt)
    __builtin_amdgcn_s_wait_asynccnt(0);
#else
    asm volatile("s_wait_asynccnt 0x0" ::: "memory");
#endif
}

union FragAB { v16bf v; uint4 q[2]; };

// ---------------------------------------------------------------------------
// Weight fp32 -> bf16, pre-swizzled into WMMA B-fragment layout:
//   dst[((kb*48 + nt)*32 + lane)*16 + j]
// lane<16:  j=0..7 -> K=kb*32+j,      j=8..15 -> K=kb*32+16+(j-8), N = nt*16+lane
// lane>=16: j=0..7 -> K=kb*32+8+j,    j=8..15 -> K=kb*32+24+(j-8), N = nt*16+(lane-16)
// ---------------------------------------------------------------------------
__global__ void convert_w_kernel(const float* __restrict__ w, u16* __restrict__ wsw) {
    const int tid = blockIdx.x * blockDim.x + threadIdx.x;
    if (tid >= 24 * 48 * 32) return;
    const int lane = tid & 31;
    const int nt   = (tid >> 5) % 48;
    const int kb   = tid / (32 * 48);
    const int col  = nt * 16 + (lane & 15);
    const int k0   = kb * 32 + ((lane < 16) ? 0  : 8);
    const int k1   = kb * 32 + ((lane < 16) ? 16 : 24);
    alignas(16) u16 vals[16];
#pragma unroll
    for (int j = 0; j < 8; ++j) vals[j]     = f2bf(w[(size_t)(k0 + j) * D + col]);
#pragma unroll
    for (int j = 0; j < 8; ++j) vals[8 + j] = f2bf(w[(size_t)(k1 + j) * D + col]);
    uint4* dst = reinterpret_cast<uint4*>(wsw + (size_t)tid * 16);
    dst[0] = *reinterpret_cast<const uint4*>(&vals[0]);
    dst[1] = *reinterpret_cast<const uint4*>(&vals[8]);
}

// ---------------------------------------------------------------------------
// GEMM helpers: 64 rows (4 M-tiles) x 768 cols (48 N-tiles, 3 per wave), K=768.
// ---------------------------------------------------------------------------
__device__ __forceinline__ void gemm_tiles(const u16 (&sA)[ROWS][AST],
                                           const u16* __restrict__ wsw,
                                           int wave, int lane, v8f acc[MT][NPW]) {
    const v8f vzero = {0.f, 0.f, 0.f, 0.f, 0.f, 0.f, 0.f, 0.f};
#pragma unroll
    for (int m = 0; m < MT; ++m)
#pragma unroll
        for (int n = 0; n < NPW; ++n) acc[m][n] = vzero;

    const int rowA = lane & 15;
    const int kofs = (lane >> 4) << 3;   // 0 for lanes 0-15, 8 for lanes 16-31
    for (int kb = 0; kb < 24; ++kb) {
        FragAB a[MT];
#pragma unroll
        for (int m = 0; m < MT; ++m) {
            const uint4* pa = reinterpret_cast<const uint4*>(&sA[m * 16 + rowA][kb * 32 + kofs]);
            a[m].q[0] = pa[0]; a[m].q[1] = pa[2];   // +32B = +16 K positions
        }
#pragma unroll
        for (int n = 0; n < NPW; ++n) {
            const int nt = wave * NPW + n;
            FragAB b;
            const uint4* pb = reinterpret_cast<const uint4*>(
                wsw + ((size_t)(kb * 48 + nt) * 32 + lane) * 16);
            b.q[0] = pb[0]; b.q[1] = pb[1];
#pragma unroll
            for (int m = 0; m < MT; ++m)
                acc[m][n] = __builtin_amdgcn_wmma_f32_16x16x32_bf16(
                    false, a[m].v, false, b.v, (short)0, acc[m][n], false, false);
        }
    }
}

__device__ __forceinline__ void store_acc(u16 (&sA)[ROWS][AST],
                                          const float* __restrict__ bias,
                                          int wave, int lane, const v8f acc[MT][NPW]) {
    const int col16 = lane & 15;
    const int rofs  = (lane >> 4) << 3;   // C layout: VGPR r -> M = r (+8 for lanes>=16)
#pragma unroll
    for (int n = 0; n < NPW; ++n) {
        const int col = (wave * NPW + n) * 16 + col16;
        const float bv = bias[col];
#pragma unroll
        for (int m = 0; m < MT; ++m)
#pragma unroll
            for (int r = 0; r < 8; ++r)
                sA[m * 16 + rofs + r][col] = f2bf(acc[m][n][r] + bv);
    }
}

__device__ __forceinline__ void ln_inplace(u16 (&sA)[ROWS][AST], float (&red)[ROWS][8][2],
                                           const float* __restrict__ g,
                                           const float* __restrict__ be,
                                           bool relu, int tid) {
    const int r = tid >> 3, t = tid & 7;   // 8 threads per row
    float s = 0.f, ss = 0.f;
    for (int i = 0; i < 96; ++i) {
        const float v = bf2f(sA[r][t * 96 + i]);
        s += v; ss += v * v;
    }
    red[r][t][0] = s; red[r][t][1] = ss;
    __syncthreads();
    float S = 0.f, SS = 0.f;
#pragma unroll
    for (int i = 0; i < 8; ++i) { S += red[r][i][0]; SS += red[r][i][1]; }
    const float mu   = S * (1.f / 768.f);
    const float var  = SS * (1.f / 768.f) - mu * mu;
    const float rstd = rsqrtf(var + 1e-5f);
    for (int i = 0; i < 96; ++i) {
        const int c = t * 96 + i;
        float v = (bf2f(sA[r][c]) - mu) * rstd * g[c] + be[c];
        if (relu) v = fmaxf(v, 0.f);
        sA[r][c] = f2bf(v);
    }
    __syncthreads();
}

__device__ __forceinline__ void l2_inplace(u16 (&sA)[ROWS][AST], float (&red)[ROWS][8][2],
                                           int tid) {
    const int r = tid >> 3, t = tid & 7;
    float ss = 0.f;
    for (int i = 0; i < 96; ++i) { const float v = bf2f(sA[r][t * 96 + i]); ss += v * v; }
    red[r][t][0] = ss;
    __syncthreads();
    float SS = 0.f;
#pragma unroll
    for (int i = 0; i < 8; ++i) SS += red[r][i][0];
    const float sc = 1.f / fmaxf(sqrtf(SS), 1e-12f);
    for (int i = 0; i < 96; ++i) {
        const int c = t * 96 + i;
        sA[r][c] = f2bf(bf2f(sA[r][c]) * sc);
    }
    __syncthreads();
}

// ---------------------------------------------------------------------------
// Fused MLP: x @ w1 + b1 -> LN -> ReLU -> @ w2 + b2 -> LN -> l2norm, bf16 out.
// One workgroup = 64 rows. 512 threads = 16 waves.
// ---------------------------------------------------------------------------
__global__ __launch_bounds__(MLP_THREADS, 1) void mlp_ln_l2_kernel(
    const float* __restrict__ x,
    const u16* __restrict__ w1sw, const float* __restrict__ b1,
    const float* __restrict__ g1, const float* __restrict__ be1,
    const u16* __restrict__ w2sw, const float* __restrict__ b2,
    const float* __restrict__ g2, const float* __restrict__ be2,
    u16* __restrict__ out) {
    __shared__ u16   sA[ROWS][AST];      // ~99 KB (CDNA5 WGP has 320 KB LDS)
    __shared__ float red[ROWS][8][2];    // 4 KB

    const int tid  = threadIdx.x;
    const int lane = tid & 31;
    const int wave = tid >> 5;
    const int row0 = blockIdx.x * ROWS;

    // Load ROWSx768 fp32 input block, convert to bf16 into LDS.
    {
        const int r = tid >> 3, t = tid & 7;
        const float* src = x + (size_t)(row0 + r) * D + t * 96;
#pragma unroll 4
        for (int i = 0; i < 96; i += 4) {
            const float4 v = *reinterpret_cast<const float4*>(src + i);
            sA[r][t * 96 + i + 0] = f2bf(v.x);
            sA[r][t * 96 + i + 1] = f2bf(v.y);
            sA[r][t * 96 + i + 2] = f2bf(v.z);
            sA[r][t * 96 + i + 3] = f2bf(v.w);
        }
    }
    __syncthreads();

    v8f acc[MT][NPW];

    gemm_tiles(sA, w1sw, wave, lane, acc);
    __syncthreads();
    store_acc(sA, b1, wave, lane, acc);
    __syncthreads();
    ln_inplace(sA, red, g1, be1, /*relu=*/true, tid);

    gemm_tiles(sA, w2sw, wave, lane, acc);
    __syncthreads();
    store_acc(sA, b2, wave, lane, acc);
    __syncthreads();
    ln_inplace(sA, red, g2, be2, /*relu=*/false, tid);
    l2_inplace(sA, red, tid);

    // Copy out as bf16.
    {
        const int r = tid >> 3, t = tid & 7;
        const uint4* srcp = reinterpret_cast<const uint4*>(&sA[r][t * 96]);
        uint4* dstp = reinterpret_cast<uint4*>(out + (size_t)(row0 + r) * D + t * 96);
#pragma unroll
        for (int i = 0; i < 12; ++i) dstp[i] = srcp[i];
    }
}

// ---------------------------------------------------------------------------
// 6-shot pointer selection. One workgroup per batch row.
// Candidate tile staged LDS-direct via GLOBAL_LOAD_ASYNC_TO_LDS_B128.
// ---------------------------------------------------------------------------
__global__ __launch_bounds__(256, 1) void score_kernel(
    const u16* __restrict__ qn, const u16* __restrict__ cn,
    float* __restrict__ logits, float* __restrict__ preds) {
    __shared__ u16   sc[KC][D];     // 48 KB candidate tile (bf16), contiguous
    __shared__ float cq[D];
    __shared__ float scores[KC];
    __shared__ int   mask[KC];
    __shared__ int   predS;
    __shared__ float ssS;

    const int tid  = threadIdx.x;
    const int lane = tid & 31;
    const int wave = tid >> 5;
    const int b    = blockIdx.x;

    {   // async copy of the 48 KB candidate tile straight into LDS (ASYNCcnt path)
        const u64 gbase   = (u64)(uintptr_t)(cn + (size_t)b * KC * D);
        const u32 ldsbase = (u32)(uintptr_t)(&sc[0][0]);
#pragma unroll 4
        for (int i = tid; i < KC * D / 8; i += 256) {
            const u32 lds_addr = ldsbase + (u32)i * 16u;
            const u64 gaddr    = gbase + (u64)i * 16ull;
            asm volatile("global_load_async_to_lds_b128 %0, %1, off"
                         :: "v"(lds_addr), "v"(gaddr)
                         : "memory");
        }
        wait_asynccnt0();
    }
    for (int i = tid; i < D; i += 256) cq[i] = bf2f(qn[(size_t)b * D + i]);
    if (tid < KC) mask[tid] = 0;
    __syncthreads();

    for (int s = 0; s < SHOT; ++s) {
        if (tid == 0) ssS = 0.f;
        // Each wave scores 4 candidates.
#pragma unroll
        for (int q = 0; q < 4; ++q) {
            const int k = wave * 4 + q;
            float part = 0.f;
#pragma unroll 4
            for (int i = 0; i < 24; ++i) {
                const int c = lane + 32 * i;
                part += cq[c] * bf2f(sc[k][c]);
            }
#pragma unroll
            for (int off = 16; off > 0; off >>= 1) part += __shfl_down(part, off);
            if (lane == 0) scores[k] = mask[k] ? -100.f : part * (1.f / TEMP);
        }
        __syncthreads();
        if (tid < KC) {
            logits[((size_t)b * SHOT + s) * KC + tid] = scores[tid];
            float v = scores[tid]; int idx = tid;        // argmax, lowest index on tie
#pragma unroll
            for (int off = 16; off > 0; off >>= 1) {
                const float ov = __shfl_down(v, off);
                const int   oi = __shfl_down(idx, off);
                if (ov > v || (ov == v && oi < idx)) { v = ov; idx = oi; }
            }
            if (tid == 0) {
                predS = idx;
                mask[idx] = 1;
                preds[(size_t)b * SHOT + s] = (float)idx;
            }
        }
        __syncthreads();
        const int p = predS;
        float vloc[3]; float part = 0.f;
#pragma unroll
        for (int j = 0; j < 3; ++j) {
            const int i = tid + 256 * j;
            const float v = 0.5f * (cq[i] + bf2f(sc[p][i]));
            vloc[j] = v; part += v * v;
        }
        atomicAdd(&ssS, part);
        __syncthreads();
        const float scale = 1.f / fmaxf(sqrtf(ssS), 1e-12f);
#pragma unroll
        for (int j = 0; j < 3; ++j) cq[tid + 256 * j] = vloc[j] * scale;
        __syncthreads();
    }
}

// ---------------------------------------------------------------------------
extern "C" void kernel_launch(void* const* d_in, const int* in_sizes, int n_in,
                              void* d_out, int out_size, void* d_ws, size_t ws_size,
                              hipStream_t stream) {
    (void)in_sizes; (void)n_in; (void)out_size; (void)ws_size;
    const float* query = (const float*)d_in[0];
    const float* cand  = (const float*)d_in[1];
    const float* qw1 = (const float*)d_in[2];
    const float* qb1 = (const float*)d_in[3];
    const float* qg1 = (const float*)d_in[4];
    const float* qbe1= (const float*)d_in[5];
    const float* qw2 = (const float*)d_in[6];
    const float* qb2 = (const float*)d_in[7];
    const float* qg2 = (const float*)d_in[8];
    const float* qbe2= (const float*)d_in[9];
    const float* cw1 = (const float*)d_in[10];
    const float* cb1 = (const float*)d_in[11];
    const float* cg1 = (const float*)d_in[12];
    const float* cbe1= (const float*)d_in[13];
    const float* cw2 = (const float*)d_in[14];
    const float* cb2 = (const float*)d_in[15];
    const float* cg2 = (const float*)d_in[16];
    const float* cbe2= (const float*)d_in[17];

    u16* ws = (u16*)d_ws;
    const size_t WSZ = (size_t)D * D;
    u16* qw1s = ws;
    u16* qw2s = ws + WSZ;
    u16* cw1s = ws + 2 * WSZ;
    u16* cw2s = ws + 3 * WSZ;
    u16* qn   = ws + 4 * WSZ;                 // [B, D] bf16
    u16* cn   = qn + (size_t)BATCH * D;       // [B, K, D] bf16

    float* logits = (float*)d_out;
    float* preds  = logits + (size_t)BATCH * SHOT * KC;

    convert_w_kernel<<<144, 256, 0, stream>>>(qw1, qw1s);
    convert_w_kernel<<<144, 256, 0, stream>>>(qw2, qw2s);
    convert_w_kernel<<<144, 256, 0, stream>>>(cw1, cw1s);
    convert_w_kernel<<<144, 256, 0, stream>>>(cw2, cw2s);

    mlp_ln_l2_kernel<<<BATCH / ROWS, MLP_THREADS, 0, stream>>>(
        query, qw1s, qb1, qg1, qbe1, qw2s, qb2, qg2, qbe2, qn);
    mlp_ln_l2_kernel<<<(BATCH * KC) / ROWS, MLP_THREADS, 0, stream>>>(
        cand, cw1s, cb1, cg1, cbe1, cw2s, cb2, cg2, cbe2, cn);

    score_kernel<<<BATCH, 256, 0, stream>>>(qn, cn, logits, preds);
}
